// HybridPhysicsGNN_20538533609735
// MI455X (gfx1250) — matrix operand order repaired and checked by reading, not compiled, not run on hardware
//
#include <hip/hip_runtime.h>

// ---------------------------------------------------------------------------
// GATv2 x2 + mean-pool + MLP for MI455X (gfx1250, wave32).
// Edge phase: gather/atomic bound, L2-resident feature tables (102MB < 192MB L2).
// Dense phase: 256->64 projections via v_wmma_f32_16x16x32_f16, exact 16x16 tiling.
// ---------------------------------------------------------------------------

typedef __attribute__((ext_vector_type(16))) _Float16 v16h;
typedef __attribute__((ext_vector_type(8)))  _Float16 v8h;
typedef __attribute__((ext_vector_type(8)))  float    v8f;

#define N_NODES  50000
#define N_EDGES  800000
#define E_TOT    850000      // edges + self loops
#define N_GRAPHS 64
#define HID      64
#define HEADS    4
#define HC       256         // HEADS*HID
#define NEG_SLOPE 0.2f

__device__ __forceinline__ float lrelu(float x) { return x > 0.f ? x : NEG_SLOPE * x; }

// monotonic float <-> ordered-u32 mapping for atomicMax-based segment max
__device__ __forceinline__ unsigned f2ord(float f) {
  unsigned u = __float_as_uint(f);
  return (u & 0x80000000u) ? ~u : (u | 0x80000000u);
}
__device__ __forceinline__ float ord2f(unsigned u) {
  return __uint_as_float((u & 0x80000000u) ? (u & 0x7fffffffu) : ~u);
}

// edge e -> (src,dst), with implicit self loops for e >= N_EDGES
__device__ __forceinline__ void edge_ends(int e, const int* ei, int& s, int& d) {
  if (e < N_EDGES) { s = ei[e]; d = ei[N_EDGES + e]; }
  else             { s = d = e - N_EDGES; }
}

// ---------------------------------------------------------------------------
__global__ void k_fill_u32(unsigned* p, unsigned v, long n) {
  long i = (long)blockIdx.x * blockDim.x + threadIdx.x;
  if (i < n) p[i] = v;
}

// layer1 linear: x[N,2] -> xl1/xr1 [N,256]; one thread per (node,out)
__global__ void k_lin1(const float* __restrict__ x,
                       const float* __restrict__ Wl, const float* __restrict__ bl,
                       const float* __restrict__ Wr, const float* __restrict__ br,
                       float* __restrict__ xl, float* __restrict__ xr) {
  long i = (long)blockIdx.x * blockDim.x + threadIdx.x;
  if (i >= (long)N_NODES * HC) return;
  int node = (int)(i >> 8), o = (int)(i & 255);
  float x0 = x[node * 2 + 0], x1 = x[node * 2 + 1];
  xl[i] = Wl[o * 2] * x0 + Wl[o * 2 + 1] * x1 + bl[o];
  xr[i] = Wr[o * 2] * x0 + Wr[o * 2 + 1] * x1 + br[o];
}

// layer1 edge logits: wave per edge, 8 channels/lane (lanes 0-7 = head0, ...).
__global__ void k_edge_logits1(const int* __restrict__ ei,
                               const float* __restrict__ xl, const float* __restrict__ xr,
                               const float* __restrict__ att,
                               float* __restrict__ logits, unsigned* __restrict__ maxb) {
  int t = blockIdx.x * blockDim.x + threadIdx.x;
  int e = t >> 5, lane = t & 31;
  if (e >= E_TOT) return;
  int s, d; edge_ends(e, ei, s, d);
  int ch = lane * 8;
  const float4* pl = (const float4*)(xl + (long)s * HC + ch);
  const float4* pr = (const float4*)(xr + (long)d * HC + ch);
  const float4* pa = (const float4*)(att + ch);
  float acc = 0.f;
#pragma unroll
  for (int j = 0; j < 2; ++j) {
    float4 a = pl[j], b = pr[j], w = pa[j];
    acc += w.x * lrelu(a.x + b.x) + w.y * lrelu(a.y + b.y)
         + w.z * lrelu(a.z + b.z) + w.w * lrelu(a.w + b.w);
  }
  acc += __shfl_xor(acc, 1, 32);
  acc += __shfl_xor(acc, 2, 32);
  acc += __shfl_xor(acc, 4, 32);
  if ((lane & 7) == 0) {
    int h = lane >> 3;
    logits[(long)e * HEADS + h] = acc;
    atomicMax(maxb + (long)d * HEADS + h, f2ord(acc));
  }
}

// layer1 exp + denom: thread per (edge, head)
__global__ void k_edge_exp1(const int* __restrict__ ei, float* __restrict__ logits,
                            const unsigned* __restrict__ maxb, float* __restrict__ denom) {
  long i = (long)blockIdx.x * blockDim.x + threadIdx.x;
  if (i >= (long)E_TOT * HEADS) return;
  int e = (int)(i >> 2), h = (int)(i & 3);
  int s, d; edge_ends(e, ei, s, d); (void)s;
  float m  = ord2f(maxb[(long)d * HEADS + h]);
  float ex = __expf(logits[i] - m);
  logits[i] = ex;
  atomicAdd(denom + (long)d * HEADS + h, ex);
}

// layer1 aggregation: wave per edge, 8 channels/lane, scatter-add to dst
__global__ void k_edge_agg1(const int* __restrict__ ei, const float* __restrict__ xl,
                            const float* __restrict__ alpha, const float* __restrict__ denom,
                            float* __restrict__ out) {
  int t = blockIdx.x * blockDim.x + threadIdx.x;
  int e = t >> 5, lane = t & 31;
  if (e >= E_TOT) return;
  int s, d; edge_ends(e, ei, s, d);
  int h = lane >> 3;
  float w = alpha[(long)e * HEADS + h] / denom[(long)d * HEADS + h];
  int ch = lane * 8;
  const float* sp = xl + (long)s * HC + ch;
  float* dp = out + (long)d * HC + ch;
#pragma unroll
  for (int j = 0; j < 8; ++j) atomicAdd(dp + j, sp[j] * w);
}

// bias + relu, emit f16 copy for the WMMA GEMM
__global__ void k_relu_bias1(const float* __restrict__ agg, const float* __restrict__ bias,
                             _Float16* __restrict__ hf) {
  long i = (long)blockIdx.x * blockDim.x + threadIdx.x;
  if (i >= (long)N_NODES * HC) return;
  float v = agg[i] + bias[i & 255];
  hf[i] = (_Float16)fmaxf(v, 0.f);
}

__global__ void k_f32_to_f16(const float* __restrict__ in, _Float16* __restrict__ out, long n) {
  long i = (long)blockIdx.x * blockDim.x + threadIdx.x;
  if (i < n) out[i] = (_Float16)in[i];
}

// out[m,n] = sum_k A[m,k]*W[n,k] + bias[n].  Wave per 16x16 tile, K=256 in 8 steps.
// ISA 7.12.2 16-bit A layout: lane L<16 holds M=L, K={k0..k0+7} (v0-3) and
// {k0+16..k0+23} (v4-7); lane L+16 holds K={k0+8..15, k0+24..31}.
// B layout: same K split as A; lane L<16 holds N=L, lane L+16 holds N=L.
__global__ void k_gemm2_wmma(const _Float16* __restrict__ A, const _Float16* __restrict__ W,
                             const float* __restrict__ bias, float* __restrict__ out) {
  int t = blockIdx.x * blockDim.x + threadIdx.x;
  int wave = t >> 5, lane = t & 31;
  if (wave >= (N_NODES / 16) * (HID / 16)) return;
  int nt = wave & (HID / 16 - 1);
  int mt = wave >> 2;
  int m0 = mt * 16, n0 = nt * 16;
  int arow = m0 + (lane & 15);
  int acol = (lane >> 4) << 3;         // 0 or 8
  int brow = n0 + (lane & 15);         // W row = output channel (B column)
  int bcol = (lane >> 4) << 3;         // 0 or 8
  const _Float16* ap = A + (long)arow * HC + acol;
  const _Float16* bp = W + (long)brow * HC + bcol;
  v8f c = {};
#pragma unroll
  for (int kk = 0; kk < HC / 32; ++kk) {
    v8h alo = *(const v8h*)(ap);
    v8h ahi = *(const v8h*)(ap + 16);
    v8h blo = *(const v8h*)(bp);
    v8h bhi = *(const v8h*)(bp + 16);
    v16h a = __builtin_shufflevector(alo, ahi, 0,1,2,3,4,5,6,7,8,9,10,11,12,13,14,15);
    v16h b = __builtin_shufflevector(blo, bhi, 0,1,2,3,4,5,6,7,8,9,10,11,12,13,14,15);
    c = __builtin_amdgcn_wmma_f32_16x16x32_f16(false, a, false, b, (short)0, c, false, false);
    ap += 32; bp += 32;
  }
  int col = lane & 15;
  int rbase = (lane >> 4) * 8;         // C layout: vgpr r -> M=r (lanes<16) / M=r+8
  float bv = bias[n0 + col];
#pragma unroll
  for (int r = 0; r < 8; ++r)
    out[(long)(m0 + rbase + r) * HID + n0 + col] = c[r] + bv;
}

// layer2 edge logits: wave per edge, 2 channels/lane, full-wave reduction
__global__ void k_edge_logits2(const int* __restrict__ ei,
                               const float* __restrict__ xl, const float* __restrict__ xr,
                               const float* __restrict__ att,
                               float* __restrict__ logits, unsigned* __restrict__ maxb) {
  int t = blockIdx.x * blockDim.x + threadIdx.x;
  int e = t >> 5, lane = t & 31;
  if (e >= E_TOT) return;
  int s, d; edge_ends(e, ei, s, d);
  int c0 = lane * 2;
  float2 a = *(const float2*)(xl + (long)s * HID + c0);
  float2 b = *(const float2*)(xr + (long)d * HID + c0);
  float2 w = *(const float2*)(att + c0);
  float acc = w.x * lrelu(a.x + b.x) + w.y * lrelu(a.y + b.y);
#pragma unroll
  for (int m = 1; m < 32; m <<= 1) acc += __shfl_xor(acc, m, 32);
  if (lane == 0) {
    logits[e] = acc;
    atomicMax(maxb + d, f2ord(acc));
  }
}

__global__ void k_edge_exp2(const int* __restrict__ ei, float* __restrict__ logits,
                            const unsigned* __restrict__ maxb, float* __restrict__ denom) {
  int e = blockIdx.x * blockDim.x + threadIdx.x;
  if (e >= E_TOT) return;
  int s, d; edge_ends(e, ei, s, d); (void)s;
  float ex = __expf(logits[e] - ord2f(maxb[d]));
  logits[e] = ex;
  atomicAdd(denom + d, ex);
}

__global__ void k_edge_agg2(const int* __restrict__ ei, const float* __restrict__ xl,
                            const float* __restrict__ alpha, const float* __restrict__ denom,
                            float* __restrict__ out) {
  int t = blockIdx.x * blockDim.x + threadIdx.x;
  int e = t >> 5, lane = t & 31;
  if (e >= E_TOT) return;
  int s, d; edge_ends(e, ei, s, d);
  float w = alpha[e] / denom[d];
  int c0 = lane * 2;
  const float* sp = xl + (long)s * HID + c0;
  float* dp = out + (long)d * HID + c0;
  atomicAdd(dp + 0, sp[0] * w);
  atomicAdd(dp + 1, sp[1] * w);
}

// bias2 + global mean-pool accumulation (sums + counts)
__global__ void k_pool(const float* __restrict__ h2, const float* __restrict__ bias,
                       const int* __restrict__ batch,
                       float* __restrict__ pool, float* __restrict__ cnt) {
  long i = (long)blockIdx.x * blockDim.x + threadIdx.x;
  if (i >= (long)N_NODES * HID) return;
  int node = (int)(i >> 6), ch = (int)(i & 63);
  int g = batch[node];
  atomicAdd(pool + g * HID + ch, h2[i] + bias[ch]);
  if (ch == 0) atomicAdd(cnt + g, 1.f);
}

// final MLP: one thread per graph
__global__ void k_mlp(const float* __restrict__ pool, const float* __restrict__ cnt,
                      const float* __restrict__ bk,
                      const float* __restrict__ w1, const float* __restrict__ b1,
                      const float* __restrict__ w2, const float* __restrict__ b2,
                      const float* __restrict__ w3, const float* __restrict__ b3,
                      float* __restrict__ out) {
  int g = threadIdx.x;
  if (g >= N_GRAPHS) return;
  float inv = 1.f / fmaxf(cnt[g], 1.f);
  float comb[65];
#pragma unroll
  for (int c = 0; c < 64; ++c) comb[c] = pool[g * 64 + c] * inv;
  comb[64] = bk[g];
  float a1[32];
  for (int o = 0; o < 32; ++o) {
    float s = b1[o];
    for (int i = 0; i < 65; ++i) s += w1[o * 65 + i] * comb[i];
    a1[o] = fmaxf(s, 0.f);
  }
  float a2[16];
  for (int o = 0; o < 16; ++o) {
    float s = b2[o];
    for (int i = 0; i < 32; ++i) s += w2[o * 32 + i] * a1[i];
    a2[o] = fmaxf(s, 0.f);
  }
  float s = b3[0];
  for (int i = 0; i < 16; ++i) s += w3[i] * a2[i];
  out[g] = bk[g] + s;
}

// ---------------------------------------------------------------------------
extern "C" void kernel_launch(void* const* d_in, const int* in_sizes, int n_in,
                              void* d_out, int out_size, void* d_ws, size_t ws_size,
                              hipStream_t stream) {
  const float* x     = (const float*)d_in[0];
  const int*   ei    = (const int*)d_in[1];
  const int*   batch = (const int*)d_in[2];
  const float* bk    = (const float*)d_in[3];
  const float* Wl1 = (const float*)d_in[4];  const float* bl1 = (const float*)d_in[5];
  const float* Wr1 = (const float*)d_in[6];  const float* br1 = (const float*)d_in[7];
  const float* at1 = (const float*)d_in[8];  const float* bi1 = (const float*)d_in[9];
  const float* Wl2 = (const float*)d_in[10]; const float* bl2 = (const float*)d_in[11];
  const float* Wr2 = (const float*)d_in[12]; const float* br2 = (const float*)d_in[13];
  const float* at2 = (const float*)d_in[14]; const float* bi2 = (const float*)d_in[15];
  const float* wc1 = (const float*)d_in[16]; const float* bc1 = (const float*)d_in[17];
  const float* wc2 = (const float*)d_in[18]; const float* bc2 = (const float*)d_in[19];
  const float* wc3 = (const float*)d_in[20]; const float* bc3 = (const float*)d_in[21];
  float* out = (float*)d_out;
  (void)in_sizes; (void)n_in; (void)out_size; (void)ws_size;

  // workspace layout (with reuse)
  char* ws = (char*)d_ws;
  auto alloc = [&](size_t bytes) { char* p = ws; ws += (bytes + 255) & ~(size_t)255; return p; };
  float*     xl1   = (float*)alloc((size_t)N_NODES * HC * 4);      // 51.2MB
  float*     xr1   = (float*)alloc((size_t)N_NODES * HC * 4);      // 51.2MB
  float*     hagg  = (float*)alloc((size_t)N_NODES * HC * 4);      // 51.2MB
  float*     alpha = (float*)alloc((size_t)E_TOT * HEADS * 4);     // 13.6MB (layer2 uses first E_TOT)
  unsigned*  maxb  = (unsigned*)alloc((size_t)N_NODES * HEADS * 4);
  float*     denom = (float*)alloc((size_t)N_NODES * HEADS * 4);
  _Float16*  wl2h  = (_Float16*)alloc((size_t)HID * HC * 2);
  _Float16*  wr2h  = (_Float16*)alloc((size_t)HID * HC * 2);
  float*     pool  = (float*)alloc((size_t)N_GRAPHS * HID * 4);
  float*     cnt   = (float*)alloc((size_t)N_GRAPHS * 4);
  // reused regions (lifetimes disjoint):
  _Float16* hf16 = (_Float16*)xr1;                 // xr1 dead after logits1
  float*    xl2  = xl1;                            // xl1 dead after agg1
  float*    xr2  = xl1 + (size_t)N_NODES * HID;
  float*    h2   = xl1 + 2 * (size_t)N_NODES * HID;

  const int B = 256;
  auto blocks = [](long n) { return (unsigned)((n + 255) / 256); };

  // ---- layer 1 ----
  k_fill_u32<<<blocks((long)N_NODES * HC), B, 0, stream>>>((unsigned*)hagg, 0u, (long)N_NODES * HC);
  k_fill_u32<<<blocks((long)N_NODES * HEADS), B, 0, stream>>>(maxb, 0u, (long)N_NODES * HEADS);
  k_fill_u32<<<blocks((long)N_NODES * HEADS), B, 0, stream>>>((unsigned*)denom, 0u, (long)N_NODES * HEADS);
  k_lin1<<<blocks((long)N_NODES * HC), B, 0, stream>>>(x, Wl1, bl1, Wr1, br1, xl1, xr1);
  k_edge_logits1<<<blocks((long)E_TOT * 32), B, 0, stream>>>(ei, xl1, xr1, at1, alpha, maxb);
  k_edge_exp1<<<blocks((long)E_TOT * HEADS), B, 0, stream>>>(ei, alpha, maxb, denom);
  k_edge_agg1<<<blocks((long)E_TOT * 32), B, 0, stream>>>(ei, xl1, alpha, denom, hagg);
  k_relu_bias1<<<blocks((long)N_NODES * HC), B, 0, stream>>>(hagg, bi1, hf16);

  // ---- layer 2 dense projections via WMMA ----
  k_f32_to_f16<<<blocks((long)HID * HC), B, 0, stream>>>(Wl2, wl2h, (long)HID * HC);
  k_f32_to_f16<<<blocks((long)HID * HC), B, 0, stream>>>(Wr2, wr2h, (long)HID * HC);
  long gemm_threads = (long)(N_NODES / 16) * (HID / 16) * 32;
  k_gemm2_wmma<<<blocks(gemm_threads), B, 0, stream>>>(hf16, wl2h, bl2, xl2);
  k_gemm2_wmma<<<blocks(gemm_threads), B, 0, stream>>>(hf16, wr2h, br2, xr2);

  // ---- layer 2 edge phase ----
  k_fill_u32<<<blocks((long)N_NODES), B, 0, stream>>>(maxb, 0u, (long)N_NODES);
  k_fill_u32<<<blocks((long)N_NODES), B, 0, stream>>>((unsigned*)denom, 0u, (long)N_NODES);
  k_fill_u32<<<blocks((long)N_NODES * HID), B, 0, stream>>>((unsigned*)h2, 0u, (long)N_NODES * HID);
  k_fill_u32<<<blocks((long)N_GRAPHS * HID), B, 0, stream>>>((unsigned*)pool, 0u, (long)N_GRAPHS * HID);
  k_fill_u32<<<blocks((long)N_GRAPHS), B, 0, stream>>>((unsigned*)cnt, 0u, (long)N_GRAPHS);
  k_edge_logits2<<<blocks((long)E_TOT * 32), B, 0, stream>>>(ei, xl2, xr2, at2, alpha, maxb);
  k_edge_exp2<<<blocks((long)E_TOT), B, 0, stream>>>(ei, alpha, maxb, denom);
  k_edge_agg2<<<blocks((long)E_TOT * 32), B, 0, stream>>>(ei, xl2, alpha, denom, h2);

  // ---- pool + MLP ----
  k_pool<<<blocks((long)N_NODES * HID), B, 0, stream>>>(h2, bi2, batch, pool, cnt);
  k_mlp<<<1, 64, 0, stream>>>(pool, cnt, bk, wc1, bc1, wc2, bc2, wc3, bc3, out);
}
